// NetworkPoissonMLP_22179211117456
// MI455X (gfx1250) — matrix operand order repaired and checked by reading, not compiled, not run on hardware
//
#include <hip/hip_runtime.h>
#include <math.h>

typedef __attribute__((ext_vector_type(16))) _Float16 v16h;
typedef __attribute__((ext_vector_type(8)))  float    v8f;

#define THETA   10.0f
#define TAU_SR  10.0f
#define K_LEN   64
#define NB      128
#define NIN     784
#define NHID    410
#define NOUT    10
#define TT      300
#define TP      304          // padded T (19 * 16)
#define NTILT   19           // T tiles per sample
#define NTILES  (NB * NTILT) // 2432 N-tiles
#define KP1     800          // padded K1
#define KT1     25           // K1 = 800 = 25*32
#define MT1     26           // M1 tiles: 416/16
#define KT2     13           // K2 = 416 = 13*32

#define WMMA_F16(a, b, c) \
    __builtin_amdgcn_wmma_f32_16x16x32_f16(false, (a), false, (b), (short)0, (c), false, false)

// A-matrix (16x32 f16) per-lane K index for half-slot j (contiguous: k = 16*(lane/16) + j)
static __device__ __forceinline__ int a_kk(int lane, int j) {
    return ((lane >= 16) ? 16 : 0) + j;
}

// ---------------- weight pre-swizzle into WMMA A-fragment layout ----------------
__global__ void prep_w1_kernel(const float* __restrict__ W1, _Float16* __restrict__ w1f) {
    int idx  = blockIdx.x * blockDim.x + threadIdx.x;   // < 26*25*512 = 332800
    int j    = idx & 15;
    int lane = (idx >> 4) & 31;
    int rest = idx >> 9;          // mt*25 + kt
    int kt   = rest % KT1;
    int mt   = rest / KT1;
    int m    = mt * 16 + (lane & 15);
    int k    = kt * 32 + a_kk(lane, j);
    _Float16 v = (_Float16)0.0f;
    if (m < NHID && k < NIN) v = (_Float16)W1[(size_t)m * NIN + k];
    w1f[idx] = v;
}

__global__ void prep_w2_kernel(const float* __restrict__ W2, _Float16* __restrict__ w2f) {
    int idx  = blockIdx.x * blockDim.x + threadIdx.x;   // < 13*512 = 6656
    int j    = idx & 15;
    int lane = (idx >> 4) & 31;
    int kt   = idx >> 9;
    int m    = lane & 15;
    int k    = kt * 32 + a_kk(lane, j);
    _Float16 v = (_Float16)0.0f;
    if (m < NOUT && k < NHID) v = (_Float16)W2[(size_t)m * NHID + k];
    w2f[idx] = v;
}

// ---------------- GEMM1: a1[o][n][t] = W1 x poisson_encode(rand_u, img) --------
__device__ __forceinline__ void store_tile(float* __restrict__ a1, v8f acc,
                                           int mt, int n, int t, int mrow) {
#pragma unroll
    for (int r = 0; r < 8; ++r) {
        int o = mt * 16 + r + mrow;
        a1[((size_t)o * NB + n) * TP + t] = acc[r];
    }
}

__global__ void gemm1_kernel(const float* __restrict__ img, const float* __restrict__ ru,
                             const _Float16* __restrict__ w1f, float* __restrict__ a1) {
    __shared__ __align__(32) _Float16 ldsb[2][512];  // double-buffered 32x16 B tile
    __shared__ float ldsimg[KP1];                    // pixel intensities for this sample
    const int lane = threadIdx.x & 31;
    const int wid  = threadIdx.x >> 5;               // 8 waves
    const int n    = blockIdx.x / NTILT;
    const int t0   = (blockIdx.x % NTILT) * 16;

    // stage pixel row once (read-only afterwards)
    for (int k = threadIdx.x; k < KP1; k += 256)
        ldsimg[k] = (k < NIN) ? img[(size_t)n * NIN + k] : 0.0f;

    // fixed per-thread encode coordinates: element e = tid + rr*256
    //   lane slot l = tid & 31, half slot j = wid + rr*8
    const int   et    = t0 + (lane & 15);
    const int   etc   = (et < TT) ? et : (TT - 1);   // clamped (always in-bounds load)
    const bool  tok   = (et < TT);
    const int   ekoff = (lane >= 16) ? 16 : 0;
    const float* ru_n = ru + (size_t)n * NIN * TT;

    v8f acc0 = {}, acc1 = {}, acc2 = {}, acc3 = {};
    const size_t mstride = (size_t)KT1 * 512;
    const size_t m3 = (wid < 2) ? (size_t)(wid + 24) : (size_t)0;  // safe addr; unused if wid>=2

    auto encode = [&](int kt, int buf) {
#pragma unroll
        for (int rr = 0; rr < 2; ++rr) {
            int   j  = wid + rr * 8;
            int   k  = kt * 32 + ekoff + j;
            int   kc = (k < NIN) ? k : (NIN - 1);            // clamped address
            float r  = ru_n[(size_t)kc * TT + etc];          // unconditional load
            float p  = ldsimg[kc];
            float v  = (r < p && k < NIN && tok) ? 1.0f : 0.0f;  // branchless select
            ldsb[buf][lane * 16 + j] = (_Float16)v;
        }
    };

    __syncthreads();          // ldsimg visible
    encode(0, 0);
    __syncthreads();          // first B tile visible

    for (int kt = 0; kt < KT1; ++kt) {
        // issue all A-fragment loads first (one clause, staggered waits)
        const size_t abase = (size_t)kt * 512 + (size_t)lane * 16;
        const v16h fa0 = *(const v16h*)&w1f[(size_t)(wid)      * mstride + abase];
        const v16h fa1 = *(const v16h*)&w1f[(size_t)(wid + 8)  * mstride + abase];
        const v16h fa2 = *(const v16h*)&w1f[(size_t)(wid + 16) * mstride + abase];
        const v16h fa3 = *(const v16h*)&w1f[m3 * mstride + abase];
        const v16h b   = *(const v16h*)&ldsb[kt & 1][lane * 16];

        // overlap: encode next tile into the other buffer while WMMAs run
        if (kt + 1 < KT1) encode(kt + 1, (kt + 1) & 1);

        acc0 = WMMA_F16(fa0, b, acc0);
        acc1 = WMMA_F16(fa1, b, acc1);
        acc2 = WMMA_F16(fa2, b, acc2);
        acc3 = WMMA_F16(fa3, b, acc3);

        __syncthreads();      // next tile written & this tile fully consumed
    }

    const int t    = t0 + (lane & 15);
    const int mrow = (lane >= 16) ? 8 : 0;
    store_tile(a1, acc0, wid,      n, t, mrow);
    store_tile(a1, acc1, wid + 8,  n, t, mrow);
    store_tile(a1, acc2, wid + 16, n, t, mrow);
    if (wid < 2) store_tile(a1, acc3, wid + 24, n, t, mrow);
}

// ---------------- psp1: SRM FIR + spike -> s1 (fp16, padded, zero-filled) ------
__global__ void psp1_kernel(const float* __restrict__ a1, _Float16* __restrict__ s1) {
    __shared__ float row[TP];
    __shared__ float eps[K_LEN];
    const int h   = blockIdx.x / NB;   // 0..415
    const int n   = blockIdx.x % NB;
    const int tid = threadIdx.x;       // blockDim = 320
    if (tid < K_LEN) {
        float tk = (float)tid;
        eps[tid] = (tk / TAU_SR) * expf(1.0f - tk / TAU_SR);
    }
    if (tid < TP) row[tid] = a1[((size_t)h * NB + n) * TP + tid];
    __syncthreads();
    if (tid < TP) {
        _Float16 s = (_Float16)0.0f;
        if (tid < TT && h < NHID) {
            float u = 0.0f;
            int kmax = tid < (K_LEN - 1) ? tid : (K_LEN - 1);
            for (int k = 0; k <= kmax; ++k) u += eps[k] * row[tid - k];
            s = (u >= THETA) ? (_Float16)1.0f : (_Float16)0.0f;
        }
        s1[((size_t)h * NB + n) * TP + tid] = s;
    }
}

// ---------------- GEMM2: a2 = W2 x s1 (one 16-row M tile) ----------------------
__global__ void gemm2_kernel(const _Float16* __restrict__ w2f, const _Float16* __restrict__ s1,
                             float* __restrict__ a2) {
    const int lane  = threadIdx.x & 31;
    const int wid   = threadIdx.x >> 5;
    const int ntile = blockIdx.x * 8 + wid;       // 304*8 = 2432 exactly
    const int n     = ntile / NTILT;
    const int t0    = (ntile % NTILT) * 16;
    const int t     = t0 + (lane & 15);

    v8f acc = {};
    for (int kt = 0; kt < KT2; ++kt) {
        const v16h a = *(const v16h*)&w2f[(size_t)kt * 512 + (size_t)lane * 16];
        const int  kb = kt * 32 + ((lane >= 16) ? 16 : 0);
        v16h b;
#pragma unroll
        for (int j = 0; j < 16; ++j)
            b[j] = s1[((size_t)(kb + j) * NB + n) * TP + t];
        acc = WMMA_F16(a, b, acc);
    }
    const int mrow = (lane >= 16) ? 8 : 0;
#pragma unroll
    for (int r = 0; r < 8; ++r)
        a2[((size_t)(r + mrow) * NB + n) * TP + t] = acc[r];
}

// ---------------- psp2: FIR + spike -> final output [n][o][t] f32 --------------
__global__ void psp2_kernel(const float* __restrict__ a2, float* __restrict__ out) {
    __shared__ float row[TP];
    __shared__ float eps[K_LEN];
    const int o   = blockIdx.x / NB;   // 0..9
    const int n   = blockIdx.x % NB;
    const int tid = threadIdx.x;
    if (tid < K_LEN) {
        float tk = (float)tid;
        eps[tid] = (tk / TAU_SR) * expf(1.0f - tk / TAU_SR);
    }
    if (tid < TP) row[tid] = a2[((size_t)o * NB + n) * TP + tid];
    __syncthreads();
    if (tid < TT) {
        float u = 0.0f;
        int kmax = tid < (K_LEN - 1) ? tid : (K_LEN - 1);
        for (int k = 0; k <= kmax; ++k) u += eps[k] * row[tid - k];
        out[((size_t)n * NOUT + o) * TT + tid] = (u >= THETA) ? 1.0f : 0.0f;
    }
}

extern "C" void kernel_launch(void* const* d_in, const int* in_sizes, int n_in,
                              void* d_out, int out_size, void* d_ws, size_t ws_size,
                              hipStream_t stream) {
    const float* img = (const float*)d_in[0];   // [128, 784]
    const float* ru  = (const float*)d_in[1];   // [128, 784, 300]
    const float* W1  = (const float*)d_in[2];   // [410, 784]
    const float* W2  = (const float*)d_in[3];   // [10, 410]
    float* out = (float*)d_out;                 // [128, 10, 300]

    char* w = (char*)d_ws;
    _Float16* w1f = (_Float16*)(w);                    //   665,600 B : W1 fragments [26][25][32][16]
    _Float16* w2f = (_Float16*)(w + 665600);           //    13,312 B : W2 fragments [13][32][16]
    float*    a1  = (float*)   (w + 678912);           // 64,749,568 B : [416][128][304] f32
    _Float16* s1  = (_Float16*)(w + 65428480);         // 32,374,784 B : [416][128][304] f16
    float*    a2  = (float*)   (w + 97803264);         //  2,490,368 B : [16][128][304] f32

    prep_w1_kernel<<<1300, 256, 0, stream>>>(W1, w1f);     // 332800 elems
    prep_w2_kernel<<<26, 256, 0, stream>>>(W2, w2f);       //   6656 elems
    gemm1_kernel<<<NTILES, 256, 0, stream>>>(img, ru, w1f, a1);
    psp1_kernel<<<416 * NB, 320, 0, stream>>>(a1, s1);
    gemm2_kernel<<<NTILES / 8, 256, 0, stream>>>(w2f, s1, a2);
    psp2_kernel<<<NOUT * NB, 320, 0, stream>>>(a2, out);
}
